// AttentionBlock_55336358643305
// MI455X (gfx1250) — matrix operand order repaired and checked by reading, not compiled, not run on hardware
//
#include <hip/hip_runtime.h>
#include <hip/hip_bf16.h>

#define B_  4
#define C_  1024
#define T_  2048
#define H_  16
#define CH_ 64
#define G_  32

typedef __attribute__((ext_vector_type(16))) __bf16 v16bf;
typedef __attribute__((ext_vector_type(8)))  float  v8f;

__device__ __forceinline__ unsigned short f2bf(float f) {
  unsigned int u = __float_as_uint(f);
  u += 0x7fffu + ((u >> 16) & 1u);           // round-to-nearest-even
  return (unsigned short)(u >> 16);
}

union FragAB { v16bf v; unsigned int u[8]; };

// A-fragment 16x32 bf16 (M x K), A row-major [M][lda], tile at (m0, kb).
__device__ __forceinline__ v16bf load_a16x32(const unsigned short* __restrict__ A,
                                             int lda, int m0, int kb) {
  int lane = threadIdx.x & 31;
  int m    = m0 + (lane & 15);
  int half = lane >> 4;
  FragAB r;
  const unsigned short* p = A + (size_t)m * lda + kb;
#pragma unroll
  for (int i = 0; i < 8; ++i) {
    int k0 = (half << 4) + (i << 1);
    r.u[i] = *reinterpret_cast<const unsigned int*>(p + k0);
  }
  return r.v;
}

// B-fragment 32x16 bf16 (K x N), given B^T row-major [N][ldb], tile at (kb, n0).
__device__ __forceinline__ v16bf load_bt32x16(const unsigned short* __restrict__ Bt,
                                              int ldb, int n0, int kb) {
  int lane = threadIdx.x & 31;
  int n    = n0 + (lane & 15);
  int half = lane >> 4;
  FragAB r;
  const unsigned int* p =
      reinterpret_cast<const unsigned int*>(Bt + (size_t)n * ldb + kb + (half << 4));
#pragma unroll
  for (int i = 0; i < 8; ++i) r.u[i] = p[i];
  return r.v;
}

__device__ __forceinline__ v8f wmma_bf16f32(v16bf a, v16bf b, v8f c) {
  return __builtin_amdgcn_wmma_f32_16x16x32_bf16(false, a, false, b, (short)0, c,
                                                 false, false);
}

// 32(M) x 64(N) wave tile, K-loop software-pipelined with ping-pong fragments.
__device__ __forceinline__ void gemm32x64(const unsigned short* __restrict__ A, int lda,
                                          const unsigned short* __restrict__ Bt, int ldb,
                                          int o0, int t0, int K, v8f acc[8]) {
  v16bf a0p, a1p, bp[4], a0q, a1q, bq[4];
  a0p = load_a16x32(A, lda, o0, 0);
  a1p = load_a16x32(A, lda, o0 + 16, 0);
#pragma unroll
  for (int nt = 0; nt < 4; ++nt) bp[nt] = load_bt32x16(Bt, ldb, t0 + nt * 16, 0);

  for (int kb = 0; kb < K; kb += 64) {
    a0q = load_a16x32(A, lda, o0, kb + 32);
    a1q = load_a16x32(A, lda, o0 + 16, kb + 32);
#pragma unroll
    for (int nt = 0; nt < 4; ++nt) bq[nt] = load_bt32x16(Bt, ldb, t0 + nt * 16, kb + 32);
#pragma unroll
    for (int nt = 0; nt < 4; ++nt) acc[nt]     = wmma_bf16f32(a0p, bp[nt], acc[nt]);
#pragma unroll
    for (int nt = 0; nt < 4; ++nt) acc[4 + nt] = wmma_bf16f32(a1p, bp[nt], acc[4 + nt]);
    if (kb + 64 < K) {
      a0p = load_a16x32(A, lda, o0, kb + 64);
      a1p = load_a16x32(A, lda, o0 + 16, kb + 64);
#pragma unroll
      for (int nt = 0; nt < 4; ++nt)
        bp[nt] = load_bt32x16(Bt, ldb, t0 + nt * 16, kb + 64);
    }
#pragma unroll
    for (int nt = 0; nt < 4; ++nt) acc[nt]     = wmma_bf16f32(a0q, bq[nt], acc[nt]);
#pragma unroll
    for (int nt = 0; nt < 4; ++nt) acc[4 + nt] = wmma_bf16f32(a1q, bq[nt], acc[4 + nt]);
  }
}

// ---------------------------------------------------------------- weights->bf16
__global__ __launch_bounds__(256) void cvt_w_kernel(const float* __restrict__ wq,
                                                    const float* __restrict__ wp,
                                                    unsigned short* __restrict__ wqb,
                                                    unsigned short* __restrict__ wpb) {
  size_t i  = (size_t)blockIdx.x * 256 + threadIdx.x;
  const size_t n1 = (size_t)3 * C_ * C_;
  if (i < n1) wqb[i] = f2bf(wq[i]);
  else        wpb[i - n1] = f2bf(wp[i - n1]);
}

// ---------------------------------------------------------------- mask -> bits
// bits[t][w] bit i  =  mask[t][w*32+i] != 0
__global__ __launch_bounds__(256) void mask_pack_kernel(
    const unsigned char* __restrict__ mask, unsigned int* __restrict__ bits) {
  int i = blockIdx.x * 256 + threadIdx.x;          // word index < T*T/32
  const uint4* p4 = reinterpret_cast<const uint4*>(mask + (size_t)i * 32);
  uint4 a = p4[0], b = p4[1];
  unsigned int words[8] = {a.x, a.y, a.z, a.w, b.x, b.y, b.z, b.w};
  unsigned int w = 0;
#pragma unroll
  for (int j = 0; j < 8; ++j) {
    unsigned int v = words[j];
    w |= ((v & 0x000000ffu) ? 1u : 0u) << (j * 4 + 0);
    w |= ((v & 0x0000ff00u) ? 1u : 0u) << (j * 4 + 1);
    w |= ((v & 0x00ff0000u) ? 1u : 0u) << (j * 4 + 2);
    w |= ((v & 0xff000000u) ? 1u : 0u) << (j * 4 + 3);
  }
  bits[i] = w;
}

// ---------------------------------------------------------------- groupnorm stats
__global__ __launch_bounds__(256) void gn_stats_kernel(const float* __restrict__ x,
                                                       float* __restrict__ stats) {
  int bg = blockIdx.x;                 // b*32 + g
  int b  = bg >> 5;
  int g  = bg & 31;
  const float4* base = reinterpret_cast<const float4*>(
      x + ((size_t)b * C_ + (size_t)g * 32) * T_);
  float s = 0.f, s2 = 0.f;
  for (int i = threadIdx.x; i < (32 * T_) / 4; i += 256) {
    float4 v = base[i];
    s  += v.x + v.y + v.z + v.w;
    s2 += v.x * v.x + v.y * v.y + v.z * v.z + v.w * v.w;
  }
  __shared__ float sh0[256], sh1[256];
  sh0[threadIdx.x] = s; sh1[threadIdx.x] = s2;
  __syncthreads();
  for (int st = 128; st > 0; st >>= 1) {
    if (threadIdx.x < st) {
      sh0[threadIdx.x] += sh0[threadIdx.x + st];
      sh1[threadIdx.x] += sh1[threadIdx.x + st];
    }
    __syncthreads();
  }
  if (threadIdx.x == 0) {
    const float inv = 1.f / (32.f * T_);
    float mu  = sh0[0] * inv;
    float var = sh1[0] * inv - mu * mu;
    stats[bg * 2]     = mu;
    stats[bg * 2 + 1] = rsqrtf(var + 1e-5f);
  }
}

// ------------------------------------------------- groupnorm apply + transpose
__global__ __launch_bounds__(256) void gn_apply_kernel(const float* __restrict__ x,
                                                       const float* __restrict__ stats,
                                                       const float* __restrict__ w,
                                                       const float* __restrict__ bia,
                                                       float* __restrict__ xn,
                                                       unsigned short* __restrict__ xnT) {
  __shared__ float tile[32][33];
  int t0 = blockIdx.x * 32, c0 = blockIdx.y * 32, b = blockIdx.z;
  int tx = threadIdx.x & 31, ty = threadIdx.x >> 5;
#pragma unroll
  for (int j = 0; j < 4; ++j) {
    int c  = c0 + ty + j * 8;
    int bg = b * 32 + (c >> 5);
    float mu = stats[bg * 2], rs = stats[bg * 2 + 1];
    size_t idx = ((size_t)b * C_ + c) * T_ + t0 + tx;
    float vv = (x[idx] - mu) * rs * w[c] + bia[c];
    xn[idx] = vv;
    tile[ty + j * 8][tx] = vv;
  }
  __syncthreads();
#pragma unroll
  for (int j = 0; j < 4; ++j) {
    int t = t0 + ty + j * 8;
    int c = c0 + tx;
    xnT[((size_t)b * T_ + t) * C_ + c] = f2bf(tile[tx][ty + j * 8]);
  }
}

// ---------------------------------------------------------------- QKV GEMM
__device__ __forceinline__ void qkv_store16(const v8f* acc4, int oB, int t0, int b,
                                            const float* __restrict__ bqkv,
                                            unsigned short* __restrict__ qT,
                                            unsigned short* __restrict__ kT,
                                            unsigned short* __restrict__ vC) {
  int lane = threadIdx.x & 31;
  int half = lane >> 4, n15 = lane & 15;
  int head = oB / 192;
  int rem  = oB % 192;
  int kind = rem / 64;            // 0=q, 1=k, 2=v
  int c0   = rem % 64;
  size_t hb = (size_t)b * H_ + head;
  const float scale = 0.35355339059327373f;  // 64^-0.25
#pragma unroll
  for (int nt = 0; nt < 4; ++nt) {
    int t = t0 + nt * 16 + n15;
    if (kind == 2) {
#pragma unroll
      for (int r = 0; r < 8; ++r) {
        int o = oB + r + 8 * half;
        int c = c0 + r + 8 * half;
        vC[(hb * CH_ + c) * (size_t)T_ + t] = f2bf(acc4[nt][r] + bqkv[o]);
      }
    } else {
      alignas(16) unsigned short tmp[8];
#pragma unroll
      for (int r = 0; r < 8; ++r) {
        int o = oB + r + 8 * half;
        tmp[r] = f2bf((acc4[nt][r] + bqkv[o]) * scale);
      }
      unsigned short* dst = ((kind == 0) ? qT : kT) +
                            (hb * T_ + t) * (size_t)CH_ + c0 + 8 * half;
      *reinterpret_cast<uint4*>(dst) = *reinterpret_cast<const uint4*>(tmp);
    }
  }
}

__global__ __launch_bounds__(256) void qkv_gemm_kernel(
    const unsigned short* __restrict__ Wq,   // [3C][C] bf16
    const unsigned short* __restrict__ xnT,  // [B][T][C] bf16  (B^T)
    const float* __restrict__ bqkv,          // [3C]
    unsigned short* __restrict__ qT,         // [B*H][T][64] bf16 (pre-scaled)
    unsigned short* __restrict__ kT,         // [B*H][T][64] bf16 (pre-scaled)
    unsigned short* __restrict__ vC) {       // [B*H][64][T] bf16
  int b    = blockIdx.z;
  int wave = threadIdx.x >> 5;
  int o0 = blockIdx.y * 256 + wave * 32;
  int t0 = blockIdx.x * 64;
  const unsigned short* Bt = xnT + (size_t)b * T_ * C_;

  v8f acc[8];
  const v8f vzero = {0.f, 0.f, 0.f, 0.f, 0.f, 0.f, 0.f, 0.f};
#pragma unroll
  for (int i = 0; i < 8; ++i) acc[i] = vzero;

  gemm32x64(Wq, C_, Bt, C_, o0, t0, C_, acc);

  qkv_store16(acc,     o0,      t0, b, bqkv, qT, kT, vC);
  qkv_store16(acc + 4, o0 + 16, t0, b, bqkv, qT, kT, vC);
}

// ---------------------------------------------------------------- attention
// Per block: one head, 8 waves x 16 query rows. K/V chunks staged block-wide
// into LDS with GLOBAL_LOAD_ASYNC_TO_LDS_B128 (ASYNCcnt), double buffered.
__global__ __launch_bounds__(256) void attn_kernel(
    const unsigned short* __restrict__ qT,   // [BH][T][64]
    const unsigned short* __restrict__ kT,   // [BH][T][64]
    const unsigned short* __restrict__ vC,   // [BH][64][T]
    const unsigned int* __restrict__ mbits,  // [T][T/32] packed mask
    unsigned short* __restrict__ aT) {       // [B][T][C] bf16 (B^T for proj)
  int head = blockIdx.y;                     // b*H + h
  int b = head >> 4, h = head & 15;
  int wave = threadIdx.x >> 5, lane = threadIdx.x & 31;
  int half = lane >> 4, n15 = lane & 15;
  int t0 = blockIdx.x * 128 + wave * 16;

  const unsigned short* q = qT + (size_t)head * T_ * CH_;
  const unsigned short* k = kT + (size_t)head * T_ * CH_;
  const unsigned short* v = vC + (size_t)head * CH_ * T_;

  __shared__ unsigned short Kbuf[2][32][64];   // [s][c] chunk,  4KB each
  __shared__ unsigned short Vbuf[2][64][32];   // [c][s] chunk,  4KB each
  __shared__ unsigned short Plds[8][16][32];

  v16bf aq0 = load_a16x32(q, CH_, t0, 0);
  v16bf aq1 = load_a16x32(q, CH_, t0, 32);

  const v8f vzero = {0.f, 0.f, 0.f, 0.f, 0.f, 0.f, 0.f, 0.f};
  v8f accv[4];
#pragma unroll
  for (int i = 0; i < 4; ++i) accv[i] = vzero;
  float mx[8], l[8];
#pragma unroll
  for (int r = 0; r < 8; ++r) { mx[r] = -1e30f; l[r] = 0.f; }

  // async block-wide staging of one 32-column chunk of K and V
  auto stage = [&](int s0, int pb) {
    int tid = threadIdx.x;
    {
      int row = tid >> 3, seg = tid & 7;                 // 32 rows x 128B
      unsigned loff = (unsigned)(unsigned long long)&Kbuf[pb][row][seg * 8];
      const unsigned short* g = k + (size_t)(s0 + row) * CH_ + seg * 8;
      asm volatile("global_load_async_to_lds_b128 %0, %1, off"
                   :: "v"(loff), "v"(g) : "memory");
    }
    {
      int row = tid >> 2, seg = tid & 3;                 // 64 rows x 64B
      unsigned loff = (unsigned)(unsigned long long)&Vbuf[pb][row][seg * 8];
      const unsigned short* g = v + (size_t)row * T_ + s0 + seg * 8;
      asm volatile("global_load_async_to_lds_b128 %0, %1, off"
                   :: "v"(loff), "v"(g) : "memory");
    }
  };
  auto async_barrier = [&]() {
    asm volatile("s_wait_asynccnt 0x0" ::: "memory");
    __syncthreads();
  };

  // B-fragment of K chunk from LDS: Bt = Kbuf[pb] as [N=s][K=c], ld=64
  auto kfrag = [&](int pb, int n0, int kb) {
    FragAB r;
    const uint4* p4 =
        reinterpret_cast<const uint4*>(&Kbuf[pb][n0 + n15][kb + 16 * half]);
    uint4 lo = p4[0], hi = p4[1];
    r.u[0] = lo.x; r.u[1] = lo.y; r.u[2] = lo.z; r.u[3] = lo.w;
    r.u[4] = hi.x; r.u[5] = hi.y; r.u[6] = hi.z; r.u[7] = hi.w;
    return r.v;
  };
  // B-fragment of V chunk from LDS: Bt = Vbuf[pb] as [N=c][K=s], ld=32
  auto vfrag = [&](int pb, int ct) {
    FragAB r;
    const uint4* p4 =
        reinterpret_cast<const uint4*>(&Vbuf[pb][ct * 16 + n15][16 * half]);
    uint4 lo = p4[0], hi = p4[1];
    r.u[0] = lo.x; r.u[1] = lo.y; r.u[2] = lo.z; r.u[3] = lo.w;
    r.u[4] = hi.x; r.u[5] = hi.y; r.u[6] = hi.z; r.u[7] = hi.w;
    return r.v;
  };

  stage(0, 0);
  async_barrier();
  int p = 0;

  for (int s0 = 0; s0 < T_; s0 += 32) {
    if (s0 + 32 < T_) stage(s0 + 32, p ^ 1);   // overlaps this chunk's compute

    v8f S0 = vzero, S1 = vzero;
    S0 = wmma_bf16f32(aq0, kfrag(p, 0, 0),  S0);
    S0 = wmma_bf16f32(aq1, kfrag(p, 0, 32), S0);
    S1 = wmma_bf16f32(aq0, kfrag(p, 16, 0),  S1);
    S1 = wmma_bf16f32(aq1, kfrag(p, 16, 32), S1);

    float p0[8], p1[8], fs[8];
#pragma unroll
    for (int r = 0; r < 8; ++r) {
      int t = t0 + r + 8 * half;
      unsigned mb = mbits[(size_t)t * (T_ / 32) + (s0 >> 5)];
      float x0 = S0[r], x1 = S1[r];
      if (!((mb >> n15) & 1u))        x0 = -__builtin_inff();
      if (!((mb >> (16 + n15)) & 1u)) x1 = -__builtin_inff();
      float mloc = fmaxf(x0, x1);
#pragma unroll
      for (int sh = 8; sh >= 1; sh >>= 1)
        mloc = fmaxf(mloc, __shfl_xor(mloc, sh, 32));
      float mnew = fmaxf(mx[r], mloc);
      float e0 = __expf(x0 - mnew);
      float e1 = __expf(x1 - mnew);
      float rs = e0 + e1;
#pragma unroll
      for (int sh = 8; sh >= 1; sh >>= 1) rs += __shfl_xor(rs, sh, 32);
      float f = __expf(mx[r] - mnew);
      l[r] = l[r] * f + rs;
      mx[r] = mnew;
      fs[r] = f;
      p0[r] = e0; p1[r] = e1;
    }

#pragma unroll
    for (int ct = 0; ct < 4; ++ct)
#pragma unroll
      for (int r = 0; r < 8; ++r) accv[ct][r] *= fs[r];

    // stage P (C layout) into LDS, re-read as A-fragment (16x32, K = s chunk)
#pragma unroll
    for (int r = 0; r < 8; ++r) {
      Plds[wave][r + 8 * half][n15]      = f2bf(p0[r]);
      Plds[wave][r + 8 * half][16 + n15] = f2bf(p1[r]);
    }
    FragAB pa;
    const unsigned short* prow = &Plds[wave][n15][0];
#pragma unroll
    for (int i = 0; i < 8; ++i) {
      int k0 = (half << 4) + (i << 1);
      pa.u[i] = *reinterpret_cast<const unsigned int*>(prow + k0);
    }

#pragma unroll
    for (int ct = 0; ct < 4; ++ct)
      accv[ct] = wmma_bf16f32(pa.v, vfrag(p, ct), accv[ct]);

    async_barrier();   // next chunk staged + everyone done with buffer p
    p ^= 1;
  }

#pragma unroll
  for (int ct = 0; ct < 4; ++ct)
#pragma unroll
    for (int r = 0; r < 8; ++r) {
      int t = t0 + r + 8 * half;
      int c = h * CH_ + ct * 16 + n15;
      aT[((size_t)b * T_ + t) * C_ + c] = f2bf(accv[ct][r] / l[r]);
    }
}

// ---------------------------------------------------------------- proj + residual
__global__ __launch_bounds__(256) void proj_gemm_kernel(
    const unsigned short* __restrict__ Wp,  // [C][C] bf16
    const unsigned short* __restrict__ aT,  // [B][T][C] bf16
    const float* __restrict__ bproj,        // [C]
    const float* __restrict__ xn,           // [B][C][T] f32
    float* __restrict__ out) {              // [B][C][T] f32
  int b    = blockIdx.z;
  int wave = threadIdx.x >> 5;
  int lane = threadIdx.x & 31;
  int o0 = blockIdx.y * 256 + wave * 32;
  int t0 = blockIdx.x * 64;
  const unsigned short* Bt = aT + (size_t)b * T_ * C_;

  v8f acc[8];
  const v8f vzero = {0.f, 0.f, 0.f, 0.f, 0.f, 0.f, 0.f, 0.f};
#pragma unroll
  for (int i = 0; i < 8; ++i) acc[i] = vzero;

  gemm32x64(Wp, C_, Bt, C_, o0, t0, C_, acc);

  int half = lane >> 4, n15 = lane & 15;
#pragma unroll
  for (int sub = 0; sub < 2; ++sub) {
#pragma unroll
    for (int nt = 0; nt < 4; ++nt) {
      int t = t0 + nt * 16 + n15;
#pragma unroll
      for (int r = 0; r < 8; ++r) {
        int o = o0 + sub * 16 + r + 8 * half;
        size_t idx = ((size_t)b * C_ + o) * T_ + t;
        out[idx] = acc[sub * 4 + nt][r] + bproj[o] + xn[idx];
      }
    }
  }
}

// ---------------------------------------------------------------- launch
extern "C" void kernel_launch(void* const* d_in, const int* in_sizes, int n_in,
                              void* d_out, int out_size, void* d_ws, size_t ws_size,
                              hipStream_t stream) {
  (void)in_sizes; (void)n_in; (void)out_size; (void)ws_size;
  const float*         x     = (const float*)d_in[0];
  const unsigned char* mask  = (const unsigned char*)d_in[1];   // jax bool -> 1 byte
  const float*         gnw   = (const float*)d_in[2];
  const float*         gnb   = (const float*)d_in[3];
  const float*         Wqkv  = (const float*)d_in[4];
  const float*         bqkv  = (const float*)d_in[5];
  const float*         Wproj = (const float*)d_in[6];
  const float*         bproj = (const float*)d_in[7];
  float* out = (float*)d_out;

  char* ws = (char*)d_ws;
  size_t off = 0;
  auto take = [&](size_t bytes) -> char* {
    char* p = ws + off;
    off += (bytes + 255) & ~(size_t)255;
    return p;
  };
  float*          stats = (float*)take((size_t)B_ * G_ * 2 * sizeof(float));
  float*          xn    = (float*)take((size_t)B_ * C_ * T_ * sizeof(float));
  unsigned short* xnT   = (unsigned short*)take((size_t)B_ * T_ * C_ * 2); // reused as aT
  unsigned short* wqb   = (unsigned short*)take((size_t)3 * C_ * C_ * 2);
  unsigned short* wpb   = (unsigned short*)take((size_t)C_ * C_ * 2);
  unsigned short* qT    = (unsigned short*)take((size_t)B_ * H_ * T_ * CH_ * 2);
  unsigned short* kT    = (unsigned short*)take((size_t)B_ * H_ * T_ * CH_ * 2);
  unsigned short* vC    = (unsigned short*)take((size_t)B_ * H_ * T_ * CH_ * 2);
  unsigned int*   mbits = (unsigned int*)take((size_t)T_ * (T_ / 32) * 4);

  cvt_w_kernel<<<(3 * C_ * C_ + C_ * C_) / 256, 256, 0, stream>>>(Wqkv, Wproj, wqb, wpb);
  mask_pack_kernel<<<(T_ * T_ / 32) / 256, 256, 0, stream>>>(mask, mbits);
  gn_stats_kernel<<<B_ * G_, 256, 0, stream>>>(x, stats);
  gn_apply_kernel<<<dim3(T_ / 32, C_ / 32, B_), 256, 0, stream>>>(x, stats, gnw, gnb,
                                                                  xn, xnT);
  qkv_gemm_kernel<<<dim3(T_ / 64, (3 * C_) / 256, B_), 256, 0, stream>>>(
      wqb, xnT, bqkv, qT, kT, vC);
  attn_kernel<<<dim3(T_ / 128, B_ * H_), 256, 0, stream>>>(qT, kT, vC, mbits, xnT);
  proj_gemm_kernel<<<dim3(T_ / 64, C_ / 256, B_), 256, 0, stream>>>(
      wpb, xnT, bproj, xn, out);
}
